// nWord2Vec_87299505258821
// MI455X (gfx1250) — compile-verified, matrix operand order checked by reading, not run on hardware
//
#include <hip/hip_runtime.h>
#include <hip/hip_bf16.h>

// Problem constants (from the reference)
#define VOCAB   100000
#define DIM     128
#define BATCH   4096
#define NNEG    5
#define DINNER  170
#define NT1     11          // ceil(170/16) N-tiles for GEMM1
#define KC1     4           // 128/32 K-chunks for GEMM1
#define NT2     8           // 128/16 N-tiles for GEMM2
#define KC2     6           // 192/32 K-chunks for GEMM2 (K padded 170 -> 192)
#define XPITCH  192         // padded inner dim for GEMM2 A
#define SQRT_D  11.313708498984761f

typedef _Float16 v4h  __attribute__((ext_vector_type(4)));
typedef _Float16 v8h  __attribute__((ext_vector_type(8)));
typedef _Float16 v16h __attribute__((ext_vector_type(16)));
typedef float    v8f  __attribute__((ext_vector_type(8)));

__device__ __forceinline__ float waveReduceSum(float v) {
#pragma unroll
  for (int m = 16; m >= 1; m >>= 1) v += __shfl_xor(v, m, 32);
  return v;
}

// Packed B-operand index for V_WMMA_F32_16X16X32_F16.
// B (KxN): lane = grp*16 + n where grp selects the K half-groups
// {0..7,16..23} (grp 0) vs {8..15,24..31} (grp 1); e = half index within lane.
__device__ __forceinline__ int pkIndex(int ntile, int kcPerTile, int kc, int n, int kl) {
  int grp = (kl >> 3) & 1;
  int e   = (kl & 7) + ((kl & 16) ? 8 : 0);
  return (((ntile * kcPerTile + kc) * 32) + grp * 16 + n) * 16 + e;
}

// Async global->LDS stage of 16 bytes per active lane (GVS addressing:
// 64-bit SGPR base + per-lane 32-bit byte offset). Tracked on ASYNCcnt.
__device__ __forceinline__ void asyncCopy16B(unsigned ldsByteAddr,
                                             unsigned byteOff,
                                             const void* base) {
  asm volatile("global_load_async_to_lds_b128 %0, %1, %2"
               :
               : "v"(ldsByteAddr), "v"(byteOff), "s"((unsigned long long)base)
               : "memory");
}
__device__ __forceinline__ void asyncWaitAll() {
  asm volatile("s_wait_asynccnt 0x0" ::: "memory");
}

// ---------------------------------------------------------------------------
// Kernel 1: normalize + f16-convert + WMMA-pack all three weight matrices.
// Waves 0..175  : W_hidden row i (rows padded 170->176, zeros)
// Waves 176..351: W_gate   row i
// Waves 352..543: W_ff_out column i (K index for GEMM2, padded 170->192)
// ---------------------------------------------------------------------------
__global__ void prep_weights_kernel(const float* __restrict__ Wh,
                                    const float* __restrict__ Wg,
                                    const float* __restrict__ Wf,
                                    _Float16* __restrict__ Whpk,
                                    _Float16* __restrict__ Wgpk,
                                    _Float16* __restrict__ Wfpk) {
  int wid  = blockIdx.x * (blockDim.x >> 5) + (threadIdx.x >> 5);
  int lane = threadIdx.x & 31;

  if (wid < 352) {
    int i = wid % 176;
    bool isG = (wid >= 176);
    const float* W = isG ? Wg : Wh;
    _Float16* pk   = isG ? Wgpk : Whpk;
    float4 v = make_float4(0.f, 0.f, 0.f, 0.f);
    if (i < DINNER) v = *(const float4*)(W + i * DIM + lane * 4);
    float ss = waveReduceSum(v.x * v.x + v.y * v.y + v.z * v.z + v.w * v.w);
    float rn = (i < DINNER) ? rsqrtf(ss) : 0.f;
    int ntile = i >> 4, n = i & 15;
    float vals[4] = {v.x, v.y, v.z, v.w};
#pragma unroll
    for (int j = 0; j < 4; ++j) {
      int k = lane * 4 + j;
      pk[pkIndex(ntile, KC1, k >> 5, n, k & 31)] = (_Float16)(vals[j] * rn);
    }
  } else if (wid < 544) {
    int i = wid - 352;                 // GEMM2 K index (column of W_ff_out)
    float vals[4];
    float ss = 0.f;
#pragma unroll
    for (int j = 0; j < 4; ++j) {
      int d = lane * 4 + j;
      float x = (i < DINNER) ? Wf[d * DINNER + i] : 0.f;
      vals[j] = x;
      ss += x * x;
    }
    ss = waveReduceSum(ss);
    float rn = (i < DINNER) ? rsqrtf(ss) : 0.f;
    int kc = i >> 5, kl = i & 31;
#pragma unroll
    for (int j = 0; j < 4; ++j) {
      int d = lane * 4 + j;
      Wfpk[pkIndex(d >> 4, KC2, kc, d & 15, kl)] = (_Float16)(vals[j] * rn);
    }
  }
}

// ---------------------------------------------------------------------------
// Kernel 2: gather W_in rows at input_ids, L2-normalize, store f16 row-major.
// Also zero the K-padding columns [170,192) of xmid.
// One wave per batch element.
// ---------------------------------------------------------------------------
__global__ void gather_emb_kernel(const int* __restrict__ input_ids,
                                  const float* __restrict__ W_in,
                                  _Float16* __restrict__ embA,
                                  _Float16* __restrict__ xmid) {
  int wid  = blockIdx.x * (blockDim.x >> 5) + (threadIdx.x >> 5);
  int lane = threadIdx.x & 31;
  if (wid >= BATCH) return;
  const float* src = W_in + (long long)input_ids[wid] * DIM;
  float4 v = *(const float4*)(src + lane * 4);
  float ss = waveReduceSum(v.x * v.x + v.y * v.y + v.z * v.z + v.w * v.w);
  float rn = rsqrtf(ss);
  v4h o;
  o.x = (_Float16)(v.x * rn);
  o.y = (_Float16)(v.y * rn);
  o.z = (_Float16)(v.z * rn);
  o.w = (_Float16)(v.w * rn);
  *(v4h*)(embA + wid * DIM + lane * 4) = o;
  if (lane < XPITCH - DINNER) xmid[wid * XPITCH + DINNER + lane] = (_Float16)0.f;
}

// Load the 16x32 f16 A operand for one 16-row tile / one 32-wide K chunk.
__device__ __forceinline__ v16h loadA(const _Float16* __restrict__ row, int kc, int off1) {
  union { v16h v; v8h h[2]; } au;
  au.h[0] = *(const v8h*)(row + kc * 32 + off1);
  au.h[1] = *(const v8h*)(row + kc * 32 + off1 + 16);
  return au.v;
}

// ---------------------------------------------------------------------------
// Kernel 3: GEMM1 — h and g share the A operand; SiLU-gate epilogue -> xmid f16.
// Block = 8 waves sharing one ntile: the 8 KB B panel (Wh+Wg for this ntile)
// is staged once per block via async global->LDS, then each wave computes one
// 16x16 tile for a distinct rowtile, feeding WMMA B operands from LDS.
// Grid: 32 rowgroups x 11 ntiles = 352 blocks.
// ---------------------------------------------------------------------------
__global__ void __launch_bounds__(256) ff1_kernel(
    const _Float16* __restrict__ embA,
    const _Float16* __restrict__ Whpk,
    const _Float16* __restrict__ Wgpk,
    const float* __restrict__ hidden_scale,
    const float* __restrict__ gate_scale,
    _Float16* __restrict__ xmid) {
  __shared__ __align__(32) _Float16 shBh[KC1 * 32 * 16];   // 4 KB
  __shared__ __align__(32) _Float16 shBg[KC1 * 32 * 16];   // 4 KB

  int t       = threadIdx.x;
  int lane    = t & 31;
  int wave    = t >> 5;
  int ntile   = blockIdx.x % NT1;
  int rowtile = (blockIdx.x / NT1) * 8 + wave;

  // Stage the 4 KB + 4 KB B panels: one async b128 per thread per panel.
  {
    unsigned byteOff = (unsigned)t * 16u;
    unsigned ldsH = (unsigned)(__SIZE_TYPE__)(&shBh[0]) + (unsigned)t * 16u;
    unsigned ldsG = (unsigned)(__SIZE_TYPE__)(&shBg[0]) + (unsigned)t * 16u;
    asyncCopy16B(ldsH, byteOff, Whpk + (size_t)ntile * KC1 * 32 * 16);
    asyncCopy16B(ldsG, byteOff, Wgpk + (size_t)ntile * KC1 * 32 * 16);
    asyncWaitAll();
  }
  __syncthreads();

  const _Float16* arow = embA + (rowtile * 16 + (lane & 15)) * DIM;
  int off1 = (lane >> 4) * 8;

  v8f ch = 0.0f, cg = 0.0f;
#pragma unroll
  for (int kc = 0; kc < KC1; ++kc) {
    v16h a  = loadA(arow, kc, off1);
    v16h bh = *(const v16h*)(shBh + (kc * 32 + lane) * 16);
    v16h bg = *(const v16h*)(shBg + (kc * 32 + lane) * 16);
    ch = __builtin_amdgcn_wmma_f32_16x16x32_f16(false, a, false, bh, (short)0, ch, false, false);
    cg = __builtin_amdgcn_wmma_f32_16x16x32_f16(false, a, false, bg, (short)0, cg, false, false);
  }

  int n  = ntile * 16 + (lane & 15);
  float hs = (n < DINNER) ? hidden_scale[n] : 0.f;
  float gs = (n < DINNER) ? gate_scale[n] * SQRT_D : 0.f;
#pragma unroll
  for (int r = 0; r < 8; ++r) {
    int m = r + (lane >> 4) * 8;
    float h = ch[r] * hs;
    float g = cg[r] * gs;
    float x = h * g / (1.f + __expf(-g));     // silu(g) * h
    xmid[(rowtile * 16 + m) * XPITCH + n] = (_Float16)x;
  }
}

// ---------------------------------------------------------------------------
// Kernel 4: GEMM2 — (B x 192) @ (192 x 128) -> xout f32.
// Same structure: block = 8 waves sharing one ntile; 6 KB B panel staged
// async into LDS. Grid: 32 rowgroups x 8 ntiles = 256 blocks.
// ---------------------------------------------------------------------------
__global__ void __launch_bounds__(256) ff2_kernel(
    const _Float16* __restrict__ xmid,
    const _Float16* __restrict__ Wfpk,
    float* __restrict__ xout) {
  __shared__ __align__(32) _Float16 shB[KC2 * 32 * 16];    // 6 KB

  int t       = threadIdx.x;
  int lane    = t & 31;
  int wave    = t >> 5;
  int ntile   = blockIdx.x % NT2;
  int rowtile = (blockIdx.x / NT2) * 8 + wave;

  {
    const void* base = Wfpk + (size_t)ntile * KC2 * 32 * 16;
    unsigned ldsBase = (unsigned)(__SIZE_TYPE__)(&shB[0]);
    unsigned byteOff = (unsigned)t * 16u;
    asyncCopy16B(ldsBase + byteOff, byteOff, base);
    if (t < 128) {                       // remaining 2 KB of the 6 KB panel
      unsigned byteOff2 = 4096u + (unsigned)t * 16u;
      asyncCopy16B(ldsBase + byteOff2, byteOff2, base);
    }
    asyncWaitAll();
  }
  __syncthreads();

  const _Float16* arow = xmid + (rowtile * 16 + (lane & 15)) * XPITCH;
  int off1 = (lane >> 4) * 8;

  v8f c = 0.0f;
#pragma unroll
  for (int kc = 0; kc < KC2; ++kc) {
    v16h a = loadA(arow, kc, off1);
    v16h b = *(const v16h*)(shB + (kc * 32 + lane) * 16);
    c = __builtin_amdgcn_wmma_f32_16x16x32_f16(false, a, false, b, (short)0, c, false, false);
  }

  int n = ntile * 16 + (lane & 15);
#pragma unroll
  for (int r = 0; r < 8; ++r) {
    int m = r + (lane >> 4) * 8;
    xout[(rowtile * 16 + m) * DIM + n] = c[r];
  }
}

// ---------------------------------------------------------------------------
// Kernel 5: gathered logits + log-sigmoid loss partials. One wave per b.
// ---------------------------------------------------------------------------
__global__ void loss_kernel(const int* __restrict__ target_ids,
                            const int* __restrict__ neg_ids,
                            const float* __restrict__ W_out,
                            const float* __restrict__ logit_scale,
                            const float* __restrict__ xout,
                            float* __restrict__ pos_part,
                            float* __restrict__ neg_part) {
  int wid   = blockIdx.x * (blockDim.x >> 5) + (threadIdx.x >> 5);
  int lane  = threadIdx.x & 31;
  int wslot = threadIdx.x >> 5;
  __shared__ float sp[8], sn[8];
  if (wid >= BATCH) return;

  float4 x4 = *(const float4*)(xout + wid * DIM + lane * 4);

  int ids[6];
  ids[0] = target_ids[wid];
#pragma unroll
  for (int k = 1; k < 6; ++k) ids[k] = neg_ids[wid * NNEG + k - 1];

  float pos = 0.f, neg = 0.f;
#pragma unroll
  for (int k = 0; k < 6; ++k) {
    if (k < 5) __builtin_prefetch(W_out + (long long)ids[k + 1] * DIM + lane * 4, 0, 1);
    float4 w4 = *(const float4*)(W_out + (long long)ids[k] * DIM + lane * 4);
    float dot = waveReduceSum(w4.x * x4.x + w4.y * x4.y + w4.z * x4.z + w4.w * x4.w);
    float ss  = waveReduceSum(w4.x * w4.x + w4.y * w4.y + w4.z * w4.z + w4.w * w4.w);
    float logit = dot * rsqrtf(ss) * logit_scale[ids[k]] * SQRT_D;
    float z = (k == 0) ? logit : -logit;
    float ls = fminf(z, 0.f) - log1pf(__expf(-fabsf(z)));   // stable log_sigmoid
    if (k == 0) pos += ls; else neg += ls;
  }
  if (lane == 0) { sp[wslot] = pos; sn[wslot] = neg; }
  __syncthreads();
  if (threadIdx.x == 0) {
    float P = 0.f, N = 0.f;
#pragma unroll
    for (int i = 0; i < 8; ++i) { P += sp[i]; N += sn[i]; }
    pos_part[blockIdx.x] = P;
    neg_part[blockIdx.x] = N;
  }
}

// ---------------------------------------------------------------------------
// Kernel 6: final reduction of 512 block partials -> scalar loss.
// ---------------------------------------------------------------------------
__global__ void finalize_kernel(const float* __restrict__ pos_part,
                                const float* __restrict__ neg_part,
                                float* __restrict__ out) {
  __shared__ float sp[256], sn[256];
  int t = threadIdx.x;
  sp[t] = pos_part[t] + pos_part[t + 256];
  sn[t] = neg_part[t] + neg_part[t + 256];
  __syncthreads();
  for (int s = 128; s > 0; s >>= 1) {
    if (t < s) { sp[t] += sp[t + s]; sn[t] += sn[t + s]; }
    __syncthreads();
  }
  if (t == 0)
    out[0] = -(sp[0] / (float)BATCH + sn[0] / (float)(BATCH * NNEG));
}

// ---------------------------------------------------------------------------
extern "C" void kernel_launch(void* const* d_in, const int* in_sizes, int n_in,
                              void* d_out, int out_size, void* d_ws, size_t ws_size,
                              hipStream_t stream) {
  (void)in_sizes; (void)n_in; (void)out_size; (void)ws_size;
  const int*   input_ids    = (const int*)d_in[0];
  const int*   target_ids   = (const int*)d_in[1];
  const int*   neg_ids      = (const int*)d_in[2];
  const float* W_in         = (const float*)d_in[3];
  const float* W_out        = (const float*)d_in[4];
  const float* W_hidden     = (const float*)d_in[5];
  const float* W_gate       = (const float*)d_in[6];
  const float* W_ff_out     = (const float*)d_in[7];
  const float* hidden_scale = (const float*)d_in[8];
  const float* gate_scale   = (const float*)d_in[9];
  const float* logit_scale  = (const float*)d_in[10];

  char* ws = (char*)d_ws;
  size_t off = 0;
  auto take = [&](size_t bytes) -> char* {
    char* p = ws + off;
    off = (off + bytes + 255) & ~(size_t)255;
    return p;
  };
  _Float16* embA = (_Float16*)take((size_t)BATCH * DIM * 2);
  _Float16* Whpk = (_Float16*)take((size_t)NT1 * KC1 * 32 * 16 * 2);
  _Float16* Wgpk = (_Float16*)take((size_t)NT1 * KC1 * 32 * 16 * 2);
  _Float16* Wfpk = (_Float16*)take((size_t)NT2 * KC2 * 32 * 16 * 2);
  _Float16* xmid = (_Float16*)take((size_t)BATCH * XPITCH * 2);
  float*    xout = (float*)take((size_t)BATCH * DIM * 4);
  float* pos_part = (float*)take(512 * 4);
  float* neg_part = (float*)take(512 * 4);

  prep_weights_kernel<<<68, 256, 0, stream>>>(W_hidden, W_gate, W_ff_out, Whpk, Wgpk, Wfpk);
  gather_emb_kernel<<<512, 256, 0, stream>>>(input_ids, W_in, embA, xmid);
  ff1_kernel<<<352, 256, 0, stream>>>(embA, Whpk, Wgpk, hidden_scale, gate_scale, xmid);
  ff2_kernel<<<256, 256, 0, stream>>>(xmid, Wfpk, xout);
  loss_kernel<<<512, 256, 0, stream>>>(target_ids, neg_ids, W_out, logit_scale, xout,
                                       pos_part, neg_part);
  finalize_kernel<<<1, 256, 0, stream>>>(pos_part, neg_part, (float*)d_out);
}
